// RATSQLGraphHiddenLayer_12962211299769
// MI455X (gfx1250) — matrix operand order repaired and checked by reading, not compiled, not run on hardware
//
#include <hip/hip_runtime.h>

#define B_   2
#define L_   1024
#define H_   256
#define NH_  8
#define DH_  32
#define FF_  1024
#define NL_  2
#define REL_ 100
#define EPS_ 1e-5f
#define INV_SCALE_ 0.17677669529663687f  // 1/sqrt(32)

typedef __attribute__((ext_vector_type(16))) __bf16 v16bf;
typedef __attribute__((ext_vector_type(8)))  float  v8f;
typedef __attribute__((ext_vector_type(4)))  int    vi4;

// A/B 16-bit fragments are two contiguous 16B chunks per lane (ISA 7.12.2):
//   A (16x32): lane m=l&15, half=l>>4: k = half*8..+7 and 16+half*8..+7
//   B (32x16): lane n=l&15, half=l>>4: k = half*8..+7 and 16+half*8..+7
static __device__ __forceinline__ v16bf frag_from2(uint4 lo, uint4 hi) {
  union { uint4 u[2]; v16bf v; } x;
  x.u[0] = lo; x.u[1] = hi;
  return x.v;
}
static __device__ __forceinline__ v8f wmma_bf16(v16bf a, v16bf b, v8f c) {
  return __builtin_amdgcn_wmma_f32_16x16x32_bf16(false, a, false, b, (short)0, c,
                                                 false, false);
}

// ---------------------------------------------------------------- converts
__global__ void k_conv(const float* __restrict__ src, __bf16* __restrict__ dst, int n) {
  int t = blockIdx.x * blockDim.x + threadIdx.x;
  if (t < n) dst[t] = (__bf16)src[t];
}

// Transposed convert: src (K,N) fp32 -> dst (N,K) bf16.
__global__ void k_convT(const float* __restrict__ src, __bf16* __restrict__ dst,
                        int K, int N) {
  int t = blockIdx.x * blockDim.x + threadIdx.x;
  if (t < K * N) {
    int k = t / N, n = t % N;
    dst[(size_t)n * K + k] = (__bf16)src[t];
  }
}

// ektp: (128,32) bf16 = Ek zero-padded rows.  evt: (32,128) bf16 = Ev^T padded.
__global__ void k_prep_rel(const float* __restrict__ Ek, const float* __restrict__ Ev,
                           __bf16* __restrict__ ektp, __bf16* __restrict__ evt) {
  int t = blockIdx.x * blockDim.x + threadIdx.x;
  if (t < 128 * 32) {
    int n = t >> 5, k = t & 31;
    ektp[t] = (n < REL_) ? (__bf16)Ek[n * DH_ + k] : (__bf16)0.f;
  }
  if (t < 32 * 128) {
    int d = t >> 7, r = t & 127;
    evt[t] = (r < REL_) ? (__bf16)Ev[r * DH_ + d] : (__bf16)0.f;
  }
}

// v (B*L, H) bf16 -> vt ((b*NH+h)*DH + d, L) bf16
__global__ void k_vt(const __bf16* __restrict__ v, __bf16* __restrict__ vt) {
  int t = blockIdx.x * blockDim.x + threadIdx.x;
  if (t < B_ * L_ * H_) {
    int col = t & (H_ - 1), row = t >> 8;
    int b = row >> 10, i = row & (L_ - 1);
    int h = col >> 5, d = col & 31;
    vt[(((size_t)(b * NH_ + h)) * DH_ + d) * L_ + i] = v[t];
  }
}

// ---------------------------------------------------------------- generic GEMM
// C(M,N) = A(M,K)bf16 @ BT(N,K)bf16^T; block=128thr/4waves, tile 16(M)x128(N),
// 16x32 per wave (2 WMMA / k-step). A staged via LDS (async if available).
template <int BIAS, int RELU, int CF, int CB>
__global__ __launch_bounds__(128) void k_gemm(
    const __bf16* __restrict__ A, const __bf16* __restrict__ BT,
    const float* __restrict__ bias, float* __restrict__ Cf, __bf16* __restrict__ Cb,
    int K, int lda, int ldbt, int ldcf, int ldcb) {
  __shared__ __bf16 As[16 * 32];
  const int lane = threadIdx.x & 31, wave = threadIdx.x >> 5;
  const int m0 = blockIdx.x << 4, n0 = (blockIdx.y << 7) + (wave << 5);
  const int m = lane & 15, half = lane >> 4;
  const __bf16* Brow0 = BT + (size_t)(n0 + m) * ldbt;
  const __bf16* Brow1 = BT + (size_t)(n0 + 16 + m) * ldbt;
  v8f acc0 = {}, acc1 = {};
  for (int k0 = 0; k0 < K; k0 += 32) {
#if __has_builtin(__builtin_amdgcn_global_load_async_to_lds_b128)
    if (threadIdx.x < 64) {
      int row = threadIdx.x >> 2, col = (threadIdx.x & 3) << 3;
      __builtin_amdgcn_global_load_async_to_lds_b128(
          (__attribute__((address_space(1))) vi4*)&A[(size_t)(m0 + row) * lda + k0 + col],
          (__attribute__((address_space(3))) vi4*)&As[row * 32 + col], 0, 0);
    }
    asm volatile("s_wait_asynccnt 0x0" ::: "memory");
#else
    {
      int row = threadIdx.x >> 3, col = (threadIdx.x & 7) << 2;
      *(uint2*)&As[row * 32 + col] =
          *(const uint2*)&A[(size_t)(m0 + row) * lda + k0 + col];
    }
#endif
    __syncthreads();
    const __bf16* ar = &As[m * 32 + (half << 3)];
    v16bf a = frag_from2(*(const uint4*)ar, *(const uint4*)(ar + 16));
    const int kb = k0 + (half << 3);
    v16bf b0 = frag_from2(*(const uint4*)&Brow0[kb], *(const uint4*)&Brow0[kb + 16]);
    v16bf b1 = frag_from2(*(const uint4*)&Brow1[kb], *(const uint4*)&Brow1[kb + 16]);
    acc0 = wmma_bf16(a, b0, acc0);
    acc1 = wmma_bf16(a, b1, acc1);
    __syncthreads();
  }
  const int n = n0 + m;
#pragma unroll
  for (int r = 0; r < 8; ++r) {
    const int mr = m0 + (half << 3) + r;
    float v0 = acc0[r], v1 = acc1[r];
    if (BIAS) { v0 += bias[n]; v1 += bias[n + 16]; }
    if (RELU) { v0 = v0 > 0.f ? v0 : 0.f; v1 = v1 > 0.f ? v1 : 0.f; }
    if (CF) { Cf[(size_t)mr * ldcf + n] = v0; Cf[(size_t)mr * ldcf + n + 16] = v1; }
    if (CB) {
      Cb[(size_t)mr * ldcb + n] = (__bf16)v0;
      Cb[(size_t)mr * ldcb + n + 16] = (__bf16)v1;
    }
  }
}

// ---------------------------------------------------------------- qe = q @ Ek^T
__global__ __launch_bounds__(32) void k_qe(const __bf16* __restrict__ qb,
                                           const __bf16* __restrict__ ektp,
                                           float* __restrict__ qe) {
  const int lane = threadIdx.x, m = lane & 15, half = lane >> 4;
  const int i0 = blockIdx.x << 4, nt = blockIdx.y, z = blockIdx.z;
  const int b = z >> 3, h = z & 7;
  const __bf16* ar =
      qb + (size_t)b * L_ * H_ + h * DH_ + (size_t)(i0 + m) * H_ + (half << 3);
  v16bf a = frag_from2(*(const uint4*)ar, *(const uint4*)(ar + 16));
  const __bf16* br = ektp + (size_t)((nt << 4) + m) * 32 + (half << 3);
  v16bf bb = frag_from2(*(const uint4*)br, *(const uint4*)(br + 16));
  v8f acc = {};
  acc = wmma_bf16(a, bb, acc);
  const int rcol = (nt << 4) + m;
  if (rcol < REL_) {
#pragma unroll
    for (int r = 0; r < 8; ++r) {
      int i = i0 + (half << 3) + r;
      qe[((size_t)z * L_ + i) * REL_ + rcol] = acc[r];
    }
  }
}

// ---------------------------------------------------------------- fused attention
__global__ __launch_bounds__(32) void k_attn(
    const __bf16* __restrict__ qb, const __bf16* __restrict__ kb,
    const __bf16* __restrict__ vt, const float* __restrict__ qe,
    const __bf16* __restrict__ evt, const int* __restrict__ rel,
    const unsigned char* __restrict__ relmask, __bf16* __restrict__ ob) {
  __shared__ __bf16 sc[16 * L_];      // 32KB score stripe
  __shared__ float bins[16 * 128];    // padded relation histogram
  __shared__ float rowStat[16];
  const int lane = threadIdx.x, m = lane & 15, half = lane >> 4;
  const int i0 = blockIdx.x << 4;
  const int z = blockIdx.y, b = z >> 3, h = z & 7;
  const __bf16* Qb = qb + (size_t)b * L_ * H_ + h * DH_;
  const __bf16* Kb = kb + (size_t)b * L_ * H_ + h * DH_;
  const __bf16* vTb = vt + (size_t)z * DH_ * L_;
  const int* relb = rel + (size_t)b * L_ * L_;
  const unsigned char* mb = relmask + (size_t)b * L_ * L_;
  const float* qeb = qe + (size_t)z * L_ * REL_;

  const __bf16* qr = Qb + (size_t)(i0 + m) * H_ + (half << 3);
  const v16bf qa = frag_from2(*(const uint4*)qr, *(const uint4*)(qr + 16));

  float mx[8];
#pragma unroll
  for (int r = 0; r < 8; ++r) mx[r] = -3.0e38f;

  // ---- phase 1: scores (q@k^T WMMA + qe gather + mask), row max ----
  for (int jt = 0; jt < L_ / 16; ++jt) {
    const int j0 = jt << 4;
    const __bf16* kr = Kb + (size_t)(j0 + m) * H_ + (half << 3);
    v16bf kf = frag_from2(*(const uint4*)kr, *(const uint4*)(kr + 16));
    v8f s = {};
    s = wmma_bf16(qa, kf, s);
    const int j = j0 + m;  // C layout: column = lane&15
#pragma unroll
    for (int r = 0; r < 8; ++r) {
      const int i = i0 + (half << 3) + r;
      const int ridx = relb[(size_t)i * L_ + j];
      float sv = (s[r] + qeb[(size_t)i * REL_ + ridx]) * INV_SCALE_;
      if (mb[(size_t)i * L_ + j]) sv = -1.0e20f;
      sc[((half << 3) + r) * L_ + j] = (__bf16)sv;
      mx[r] = fmaxf(mx[r], sv);
    }
  }
#pragma unroll
  for (int r = 0; r < 8; ++r)
#pragma unroll
    for (int o2 = 8; o2 > 0; o2 >>= 1)
      mx[r] = fmaxf(mx[r], __shfl_xor(mx[r], o2, 32));
  if (m == 0) {
#pragma unroll
    for (int r = 0; r < 8; ++r) rowStat[(half << 3) + r] = mx[r];
  }
  __syncthreads();
  const float rmax = rowStat[m];  // A-frag rows use m
  for (int t = lane; t < 16 * 128; t += 32) bins[t] = 0.f;
  __syncthreads();

  // ---- phase 2: p = exp(s-max); o += p@v ; relation histogram ----
  v8f o0 = {}, o1 = {};
  float rsum = 0.f;
  for (int jt = 0; jt < L_ / 32; ++jt) {
    const int j0 = jt << 5;
    const __bf16* sr = &sc[m * L_ + j0 + (half << 3)];
    union { uint4 u[2]; __bf16 hh[16]; } sraw;
    sraw.u[0] = *(const uint4*)sr;
    sraw.u[1] = *(const uint4*)(sr + 16);
    union { __bf16 hh[16]; v16bf v; } pk;
    const size_t relrow = (size_t)(i0 + m) * L_ + j0;
    float rs = 0.f;
#pragma unroll
    for (int e = 0; e < 16; ++e) {
      const int k = ((e & 8) << 1) + (half << 3) + (e & 7);
      const float pv = __expf((float)sraw.hh[e] - rmax);
      pk.hh[e] = (__bf16)pv;
      rs += pv;
      atomicAdd(&bins[(m << 7) + relb[relrow + k]], pv);
    }
    rsum += rs;
    const __bf16* vr0 = vTb + (size_t)m * L_ + j0 + (half << 3);
    const __bf16* vr1 = vTb + (size_t)(16 + m) * L_ + j0 + (half << 3);
    v16bf v0f = frag_from2(*(const uint4*)vr0, *(const uint4*)(vr0 + 16));
    v16bf v1f = frag_from2(*(const uint4*)vr1, *(const uint4*)(vr1 + 16));
    o0 = wmma_bf16(pk.v, v0f, o0);
    o1 = wmma_bf16(pk.v, v1f, o1);
  }
  rsum += __shfl_xor(rsum, 16, 32);
  __syncthreads();  // histogram atomics done; safe to reuse rowStat
  if (half == 0) rowStat[m] = rsum;
  __syncthreads();

  // ---- phase 3: o += bins @ Ev (K padded to 128) ----
#pragma unroll
  for (int ks = 0; ks < 4; ++ks) {
    const float* br = &bins[(m << 7) + (ks << 5) + (half << 3)];
    union { float4 f4[4]; float f[16]; } bfl;
    bfl.f4[0] = *(const float4*)br;
    bfl.f4[1] = *(const float4*)(br + 4);
    bfl.f4[2] = *(const float4*)(br + 16);
    bfl.f4[3] = *(const float4*)(br + 20);
    union { __bf16 hh[16]; v16bf v; } ba;
#pragma unroll
    for (int e = 0; e < 16; ++e) ba.hh[e] = (__bf16)bfl.f[e];
    const __bf16* er0 = evt + (size_t)m * 128 + (ks << 5) + (half << 3);
    const __bf16* er1 = evt + (size_t)(16 + m) * 128 + (ks << 5) + (half << 3);
    v16bf e0 = frag_from2(*(const uint4*)er0, *(const uint4*)(er0 + 16));
    v16bf e1 = frag_from2(*(const uint4*)er1, *(const uint4*)(er1 + 16));
    o0 = wmma_bf16(ba.v, e0, o0);
    o1 = wmma_bf16(ba.v, e1, o1);
  }
#pragma unroll
  for (int r = 0; r < 8; ++r) {
    const int i = i0 + (half << 3) + r;
    const float inv = 1.0f / rowStat[(half << 3) + r];
    __bf16* orow = ob + ((size_t)b * L_ + i) * H_ + h * DH_;
    orow[m] = (__bf16)(o0[r] * inv);
    orow[16 + m] = (__bf16)(o1[r] * inv);
  }
}

// ---------------------------------------------------------------- residual + LN
__global__ __launch_bounds__(256) void k_ln(const float* __restrict__ xin,
                                            const float* __restrict__ del,
                                            const float* __restrict__ g,
                                            const float* __restrict__ be,
                                            float* __restrict__ xout,
                                            __bf16* __restrict__ xbout) {
  const int row = blockIdx.x, t = threadIdx.x;
  const size_t idx = (size_t)row * H_ + t;
  const float v = xin[idx] + del[idx];
  __shared__ float red[8];
  float s = v;
#pragma unroll
  for (int o2 = 16; o2 > 0; o2 >>= 1) s += __shfl_xor(s, o2, 32);
  if ((t & 31) == 0) red[t >> 5] = s;
  __syncthreads();
  float tot = 0.f;
#pragma unroll
  for (int i = 0; i < 8; ++i) tot += red[i];
  const float mean = tot * (1.0f / H_);
  const float d = v - mean;
  __syncthreads();
  float s2 = d * d;
#pragma unroll
  for (int o2 = 16; o2 > 0; o2 >>= 1) s2 += __shfl_xor(s2, o2, 32);
  if ((t & 31) == 0) red[t >> 5] = s2;
  __syncthreads();
  float tot2 = 0.f;
#pragma unroll
  for (int i = 0; i < 8; ++i) tot2 += red[i];
  const float var = tot2 * (1.0f / H_);
  const float y = d * rsqrtf(var + EPS_) * g[t] + be[t];
  xout[idx] = y;
  xbout[idx] = (__bf16)y;
}

// ---------------------------------------------------------------- host
extern "C" void kernel_launch(void* const* d_in, const int* in_sizes, int n_in,
                              void* d_out, int out_size, void* d_ws, size_t ws_size,
                              hipStream_t stream) {
  (void)in_sizes; (void)n_in; (void)out_size; (void)ws_size;
  const float* inputs = (const float*)d_in[0];
  const float* Wq = (const float*)d_in[1];
  const float* Wk = (const float*)d_in[2];
  const float* Wv = (const float*)d_in[3];
  const float* Wo = (const float*)d_in[4];
  const float* bo = (const float*)d_in[5];
  const float* W1 = (const float*)d_in[6];
  const float* b1 = (const float*)d_in[7];
  const float* W2 = (const float*)d_in[8];
  const float* b2 = (const float*)d_in[9];
  const float* ln1g = (const float*)d_in[10];
  const float* ln1b = (const float*)d_in[11];
  const float* ln2g = (const float*)d_in[12];
  const float* ln2b = (const float*)d_in[13];
  const float* Ek = (const float*)d_in[14];
  const float* Ev = (const float*)d_in[15];
  const int* relations = (const int*)d_in[16];
  const unsigned char* relmask = (const unsigned char*)d_in[17];
  float* out = (float*)d_out;

  char* p = (char*)d_ws;
  auto alloc = [&](size_t bytes) -> char* {
    char* r = p; p += (bytes + 255) & ~(size_t)255; return r;
  };
  const size_t ROWS = (size_t)B_ * L_;
  __bf16* xb  = (__bf16*)alloc(ROWS * H_ * 2);
  __bf16* qbf = (__bf16*)alloc(ROWS * H_ * 2);
  __bf16* kbf = (__bf16*)alloc(ROWS * H_ * 2);
  __bf16* vbf = (__bf16*)alloc(ROWS * H_ * 2);
  __bf16* obf = (__bf16*)alloc(ROWS * H_ * 2);
  __bf16* vt  = (__bf16*)alloc(ROWS * H_ * 2);
  __bf16* ffb = (__bf16*)alloc(ROWS * FF_ * 2);
  float*  yf  = (float*)alloc(ROWS * H_ * 4);
  float*  xf  = (float*)alloc(ROWS * H_ * 4);
  float*  qe  = (float*)alloc((size_t)B_ * NH_ * L_ * REL_ * 4);
  __bf16* wqt = (__bf16*)alloc((size_t)NL_ * H_ * H_ * 2);
  __bf16* wkt = (__bf16*)alloc((size_t)NL_ * H_ * H_ * 2);
  __bf16* wvt = (__bf16*)alloc((size_t)NL_ * H_ * H_ * 2);
  __bf16* wot = (__bf16*)alloc((size_t)NL_ * H_ * H_ * 2);
  __bf16* w1t = (__bf16*)alloc((size_t)NL_ * H_ * FF_ * 2);
  __bf16* w2t = (__bf16*)alloc((size_t)NL_ * FF_ * H_ * 2);
  __bf16* ektp = (__bf16*)alloc(128 * 32 * 2);
  __bf16* evt  = (__bf16*)alloc(32 * 128 * 2);

  for (int l = 0; l < NL_; ++l) {
    const int HH = H_ * H_, HF = H_ * FF_;
    k_convT<<<(HH + 255) / 256, 256, 0, stream>>>(Wq + (size_t)l * HH,
                                                  wqt + (size_t)l * HH, H_, H_);
    k_convT<<<(HH + 255) / 256, 256, 0, stream>>>(Wk + (size_t)l * HH,
                                                  wkt + (size_t)l * HH, H_, H_);
    k_convT<<<(HH + 255) / 256, 256, 0, stream>>>(Wv + (size_t)l * HH,
                                                  wvt + (size_t)l * HH, H_, H_);
    k_convT<<<(HH + 255) / 256, 256, 0, stream>>>(Wo + (size_t)l * HH,
                                                  wot + (size_t)l * HH, H_, H_);
    k_convT<<<(HF + 255) / 256, 256, 0, stream>>>(W1 + (size_t)l * HF,
                                                  w1t + (size_t)l * HF, H_, FF_);
    k_convT<<<(HF + 255) / 256, 256, 0, stream>>>(W2 + (size_t)l * HF,
                                                  w2t + (size_t)l * HF, FF_, H_);
  }
  k_conv<<<(B_ * L_ * H_ + 255) / 256, 256, 0, stream>>>(inputs, xb, B_ * L_ * H_);
  k_prep_rel<<<16, 256, 0, stream>>>(Ek, Ev, ektp, evt);

  const float* xin = inputs;
  for (int l = 0; l < NL_; ++l) {
    const size_t WHH = (size_t)l * H_ * H_;
    const size_t WHF = (size_t)l * H_ * FF_;
    dim3 gP(ROWS / 16, H_ / 128);
    k_gemm<0, 0, 0, 1><<<gP, 128, 0, stream>>>(xb, wqt + WHH, nullptr, nullptr, qbf,
                                               H_, H_, H_, 0, H_);
    k_gemm<0, 0, 0, 1><<<gP, 128, 0, stream>>>(xb, wkt + WHH, nullptr, nullptr, kbf,
                                               H_, H_, H_, 0, H_);
    k_gemm<0, 0, 0, 1><<<gP, 128, 0, stream>>>(xb, wvt + WHH, nullptr, nullptr, vbf,
                                               H_, H_, H_, 0, H_);
    k_vt<<<(B_ * L_ * H_ + 255) / 256, 256, 0, stream>>>(vbf, vt);
    k_qe<<<dim3(L_ / 16, 7, B_ * NH_), 32, 0, stream>>>(qbf, ektp, qe);
    k_attn<<<dim3(L_ / 16, B_ * NH_), 32, 0, stream>>>(qbf, kbf, vt, qe, evt,
                                                       relations, relmask, obf);
    k_gemm<1, 0, 1, 0><<<gP, 128, 0, stream>>>(obf, wot + WHH, bo + (size_t)l * H_,
                                               yf, nullptr, H_, H_, H_, H_, 0);
    k_ln<<<(int)ROWS, 256, 0, stream>>>(xin, yf, ln1g + (size_t)l * H_,
                                        ln1b + (size_t)l * H_, xf, xb);
    k_gemm<1, 1, 0, 1><<<dim3(ROWS / 16, FF_ / 128), 128, 0, stream>>>(
        xb, w1t + WHF, b1 + (size_t)l * FF_, nullptr, ffb, H_, H_, H_, 0, FF_);
    k_gemm<1, 0, 1, 0><<<gP, 128, 0, stream>>>(ffb, w2t + WHF, b2 + (size_t)l * H_,
                                               yf, nullptr, FF_, FF_, FF_, H_, 0);
    float* lnout = (l == NL_ - 1) ? out : xf;
    k_ln<<<(int)ROWS, 256, 0, stream>>>(xf, yf, ln2g + (size_t)l * H_,
                                        ln2b + (size_t)l * H_, lnout, xb);
    xin = lnout;
  }
}
